// MambaBlock3D_90194313216283
// MI455X (gfx1250) — matrix operand order repaired and checked by reading, not compiled, not run on hardware
//
#include <hip/hip_runtime.h>
#include <math.h>

// ---- problem dims (compile-time, from reference) ----
#define D_MODEL  512
#define D_STATE  64
#define D_CONV   4
#define D_INNER  1024
#define DT_RANK  32
#define SEQ      2048
#define NPROJ    (DT_RANK + 2*D_STATE)   // 160
#define CHUNK    64
#define NCHUNK   (SEQ / CHUNK)           // 32

typedef __bf16 bf16;
typedef __attribute__((ext_vector_type(16))) __bf16 v16bf;
typedef __attribute__((ext_vector_type(8)))  __bf16 v8bf;
typedef __attribute__((ext_vector_type(8)))  float  v8f;
typedef unsigned int u32x4 __attribute__((ext_vector_type(4)));
typedef int          i32x8 __attribute__((ext_vector_type(8)));
typedef int          i32x4 __attribute__((ext_vector_type(4)));

union FragAB { v16bf v; v8bf h[2]; };
union FragC  { v8f  v; float e[8]; };

__device__ __forceinline__ float sigmoidf_(float x){ return 1.f/(1.f+__expf(-x)); }
__device__ __forceinline__ float siluf_(float x){ return x*sigmoidf_(x); }
__device__ __forceinline__ float softplusf_(float x){ return (x>20.f)?x:log1pf(__expf(x)); }

// ---------------- TDM: 2D f32 tile (global -> LDS), ISA §8 descriptor ----------
// tile_w/tile_h/row_stride in elements (data_size = 4B). Issued once by one wave;
// completion via TENSORcnt. 6-arg builtin (clang-23 / therock lane).
__device__ __forceinline__ void tdm_load_2d_f32(
    unsigned lds_byte_addr, const float* gptr,
    unsigned tile_w, unsigned tile_h, unsigned row_stride)
{
  const unsigned long long ga = (unsigned long long)(size_t)gptr;
  u32x4 g0;
  g0[0] = 1u;                                   // count=1, is_restore=0, no gather
  g0[1] = lds_byte_addr;                        // D#.lds_addr
  g0[2] = (unsigned)(ga & 0xFFFFFFFFu);         // global_addr[31:0]
  g0[3] = (unsigned)((ga >> 32) & 0x01FFFFFFu)  // global_addr[56:32]
        | 0x80000000u;                          // type=2 ("image") in bits 127:126
  i32x8 g1;
  g1[0] = (int)(2u << 16);                      // data_size=2 -> 4 bytes
  g1[1] = (int)((tile_w & 0xFFFFu) << 16);      // tensor_dim0[15:0]   (== tile_w)
  g1[2] = (int)((tile_w >> 16) | ((tile_h & 0xFFFFu) << 16)); // dim0 hi | tensor_dim1 lo
  g1[3] = (int)((tile_h >> 16) | (tile_w << 16));             // dim1 hi | tile_dim0
  g1[4] = (int)tile_h;                          // tile_dim1 (tile_dim2 = 0)
  g1[5] = (int)row_stride;                      // tensor_dim0_stride[31:0]
  g1[6] = 0;                                    // stride0 hi | stride1 lo
  g1[7] = 0;                                    // stride1 hi
  const i32x4 z4 = {0,0,0,0};
  const i32x8 z8 = {0,0,0,0,0,0,0,0};
  __builtin_amdgcn_tensor_load_to_lds(g0, g1, z4, z4, z8, 0);
}

// ---------------- fp32 -> bf16 convert ----------------
__global__ void k_cvt_bf16(const float* __restrict__ src, bf16* __restrict__ dst, int n){
  int i = blockIdx.x*blockDim.x + threadIdx.x;
  if(i<n) dst[i] = (bf16)src[i];
}

// ---------------- layernorm (row=512) + bf16 ----------------
__global__ __launch_bounds__(256) void k_layernorm_bf16(
    const float* __restrict__ x, const float* __restrict__ g,
    const float* __restrict__ b, bf16* __restrict__ out){
  __shared__ float red[256];
  const int row = blockIdx.x;
  const int tid = threadIdx.x;
  const float* xr = x + (size_t)row*D_MODEL;
  float v0 = xr[tid], v1 = xr[tid+256];
  red[tid] = v0+v1; __syncthreads();
  for(int s=128;s>0;s>>=1){ if(tid<s) red[tid]+=red[tid+s]; __syncthreads(); }
  const float mu = red[0]*(1.f/D_MODEL); __syncthreads();
  const float c0=v0-mu, c1=v1-mu;
  red[tid]=c0*c0+c1*c1; __syncthreads();
  for(int s=128;s>0;s>>=1){ if(tid<s) red[tid]+=red[tid+s]; __syncthreads(); }
  const float rinv = rsqrtf(red[0]*(1.f/D_MODEL)+1e-5f);
  out[(size_t)row*D_MODEL+tid]     = (bf16)(c0*rinv*g[tid]     + b[tid]);
  out[(size_t)row*D_MODEL+tid+256] = (bf16)(c1*rinv*g[tid+256] + b[tid+256]);
}

// ---------------- bf16 WMMA GEMM: C[M,N] = A[M,K] * B[N,K]^T (+epilogue) ----
// wave tile 64x32 (4x2 wmma frags, 8 wmma / 6 fragment loads per K-step);
// 8 waves arranged 2(M) x 4(N) -> 128x128 block tile.
enum { EPI_NONE=0, EPI_SOFTPLUS_BIAS=1, EPI_RESID=2 };

template<int EPI>
__global__ __launch_bounds__(256) void k_gemm_bf16(
    const bf16* __restrict__ A, int lda,
    const bf16* __restrict__ B, int ldb,
    float* __restrict__ C, int ldc,
    int M, int N, int K,
    const float* __restrict__ aux, int ldaux)
{
  const int lane = threadIdx.x & 31;
  const int wave = threadIdx.x >> 5;
  const int m_in = lane & 15;       // row (A) / col (B,C) within 16
  const int half = lane >> 4;       // lane half select
  const int wm = blockIdx.y*128 + (wave & 1)*64;
  const int wn = blockIdx.x*128 + (wave >> 1)*32;

  FragC acc[4][2];
  #pragma unroll
  for(int i=0;i<4;i++)
    #pragma unroll
    for(int j=0;j<2;j++)
      #pragma unroll
      for(int e=0;e<8;e++) acc[i][j].e[e]=0.f;

  for(int k0=0;k0<K;k0+=32){
    FragAB a[4], bb[2];
    // A fragment (ISA 16-bit A layout): elems 0..7 -> k0+half*8+e ; 8..15 -> k0+16+half*8+e
    #pragma unroll
    for(int tm=0;tm<4;tm++){
      const bf16* ap = A + (size_t)(wm + tm*16 + m_in)*lda + k0 + half*8;
      a[tm].h[0] = *(const v8bf*)(ap);
      a[tm].h[1] = *(const v8bf*)(ap + 16);
    }
    // B fragment: lane holds column n; elems 0..7 -> k0+half*8+e ; 8..15 -> k0+16+half*8+e
    #pragma unroll
    for(int tn=0;tn<2;tn++){
      const int n  = wn + tn*16 + m_in;
      const int nc = (n < N) ? n : 0;
      const bf16* bp = B + (size_t)nc*ldb + k0 + half*8;
      bb[tn].h[0] = *(const v8bf*)(bp);
      bb[tn].h[1] = *(const v8bf*)(bp + 16);
    }
    #pragma unroll
    for(int tm=0;tm<4;tm++)
      #pragma unroll
      for(int tn=0;tn<2;tn++)
        acc[tm][tn].v = __builtin_amdgcn_wmma_f32_16x16x32_bf16(
            false, a[tm].v, false, bb[tn].v, (short)0, acc[tm][tn].v, false, false);
  }
  // C/D layout: lane -> col = m_in ; VGPR e -> row = e + 8*half
  #pragma unroll
  for(int tm=0;tm<4;tm++){
    #pragma unroll
    for(int tn=0;tn<2;tn++){
      const int n = wn + tn*16 + m_in;
      if(n < N){
        #pragma unroll
        for(int e=0;e<8;e++){
          const int m = wm + tm*16 + half*8 + e;
          float v = acc[tm][tn].e[e];
          if(EPI==EPI_SOFTPLUS_BIAS) v = softplusf_(v + aux[n]);
          if(EPI==EPI_RESID)         v = v + aux[(size_t)m*ldaux + n];
          C[(size_t)m*ldc + n] = v;
        }
      }
    }
  }
}

// ---------------- causal depthwise conv (k=4) + SiLU ----------------
__global__ void k_conv_silu(const float* __restrict__ xz, const float* __restrict__ cw,
                            const float* __restrict__ cb, float* __restrict__ xi_f,
                            bf16* __restrict__ xi_b){
  const int idx = blockIdx.x*blockDim.x + threadIdx.x;  // SEQ*D_INNER
  const int l = idx >> 10;
  const int d = idx & (D_INNER-1);
  float acc = cb[d];
  #pragma unroll
  for(int j=0;j<D_CONV;j++){
    const int t = l - (D_CONV-1) + j;
    if(t >= 0) acc += xz[(size_t)t*(2*D_INNER) + d] * cw[d*D_CONV + j];
  }
  acc = siluf_(acc);
  xi_f[idx] = acc;
  xi_b[idx] = (bf16)acc;
}

// ---------------- extract dt columns as bf16 ----------------
__global__ void k_dt_bf16(const float* __restrict__ dbc, bf16* __restrict__ dt){
  const int idx = blockIdx.x*blockDim.x + threadIdx.x;  // SEQ*DT_RANK
  const int l = idx >> 5, r = idx & (DT_RANK-1);
  dt[idx] = (bf16)dbc[(size_t)l*NPROJ + r];
}

// ---------------- chunked selective scan ----------------
// pass1: per (chunk,d): local scan with h0=0 -> hend; also sum(delta) per chunk.
// B rows (shared by all channels) staged into LDS by the Tensor Data Mover.
__global__ __launch_bounds__(256) void k_scan_pass1(
    const float* __restrict__ delta, const float* __restrict__ xi,
    const float* __restrict__ dbc,  const float* __restrict__ A_log,
    float* __restrict__ hend, float* __restrict__ sdelta)
{
  __shared__ float sB[CHUNK*D_STATE];                 // 64x64 f32 = 16KB
  const int c  = blockIdx.x & (NCHUNK-1);
  const int dg = blockIdx.x >> 5;
  const int d  = dg*256 + threadIdx.x;
  const int t0 = c*CHUNK;
  if(threadIdx.x < 32){   // wave 0 issues one TDM DMA for the whole block
    tdm_load_2d_f32((unsigned)(size_t)(void*)sB,
                    dbc + (size_t)t0*NPROJ + DT_RANK,
                    /*tile_w=*/D_STATE, /*tile_h=*/CHUNK, /*row_stride=*/NPROJ);
    __builtin_amdgcn_s_wait_tensorcnt(0);
  }
  __syncthreads();
  float Arow[D_STATE], h[D_STATE];
  #pragma unroll
  for(int n=0;n<D_STATE;n++){ Arow[n] = -__expf(A_log[(size_t)d*D_STATE+n]); h[n]=0.f; }
  float ds = 0.f;
  for(int tt=0;tt<CHUNK;tt++){
    const int t = t0+tt;
    const float dlt = delta[(size_t)t*D_INNER + d];
    const float dx  = dlt * xi[(size_t)t*D_INNER + d];
    ds += dlt;
    const float* bt = &sB[tt*D_STATE];
    #pragma unroll
    for(int n=0;n<D_STATE;n++)
      h[n] = __expf(dlt*Arow[n])*h[n] + dx*bt[n];
  }
  float* he = hend + ((size_t)c*D_INNER + d)*D_STATE;
  #pragma unroll
  for(int n=0;n<D_STATE;n++) he[n] = h[n];
  sdelta[c*D_INNER + d] = ds;
}

// pass2: serial cross-chunk combine; chunk decay factors as exp(A * sum(delta))
__global__ __launch_bounds__(256) void k_scan_pass2(
    const float* __restrict__ hend, const float* __restrict__ sdelta,
    const float* __restrict__ A_log, float* __restrict__ hstart)
{
  const int d = blockIdx.x*256 + threadIdx.x;
  float Arow[D_STATE], carry[D_STATE];
  #pragma unroll
  for(int n=0;n<D_STATE;n++){ Arow[n] = -__expf(A_log[(size_t)d*D_STATE+n]); carry[n]=0.f; }
  for(int c=0;c<NCHUNK;c++){
    const float S  = sdelta[c*D_INNER + d];
    float* hs      = hstart + ((size_t)c*D_INNER + d)*D_STATE;
    const float* he= hend   + ((size_t)c*D_INNER + d)*D_STATE;
    #pragma unroll
    for(int n=0;n<D_STATE;n++){
      hs[n]    = carry[n];
      carry[n] = he[n] + __expf(S*Arow[n])*carry[n];
    }
  }
}

// pass3: replay chunk from corrected state; fuse +Dp*xi, silu(z) gate, bf16 out.
// B and C rows (contiguous 128 cols of dbc) staged via one TDM DMA.
__global__ __launch_bounds__(256) void k_scan_pass3(
    const float* __restrict__ delta, const float* __restrict__ xi,
    const float* __restrict__ dbc,  const float* __restrict__ A_log,
    const float* __restrict__ hstart, const float* __restrict__ Dp,
    const float* __restrict__ xz, bf16* __restrict__ y_b)
{
  __shared__ float sBC[CHUNK*2*D_STATE];              // 64x128 f32 = 32KB
  const int c  = blockIdx.x & (NCHUNK-1);
  const int dg = blockIdx.x >> 5;
  const int d  = dg*256 + threadIdx.x;
  const int t0 = c*CHUNK;
  if(threadIdx.x < 32){
    tdm_load_2d_f32((unsigned)(size_t)(void*)sBC,
                    dbc + (size_t)t0*NPROJ + DT_RANK,
                    /*tile_w=*/2*D_STATE, /*tile_h=*/CHUNK, /*row_stride=*/NPROJ);
    __builtin_amdgcn_s_wait_tensorcnt(0);
  }
  __syncthreads();
  float Arow[D_STATE], h[D_STATE];
  const float* hs = hstart + ((size_t)c*D_INNER + d)*D_STATE;
  #pragma unroll
  for(int n=0;n<D_STATE;n++){ Arow[n] = -__expf(A_log[(size_t)d*D_STATE+n]); h[n]=hs[n]; }
  const float dpd = Dp[d];
  for(int tt=0;tt<CHUNK;tt++){
    const int t = t0+tt;
    const float dlt = delta[(size_t)t*D_INNER + d];
    const float xv  = xi[(size_t)t*D_INNER + d];
    const float dx  = dlt*xv;
    const float* bt = &sBC[tt*2*D_STATE];
    const float* ct = bt + D_STATE;
    float y = 0.f;
    #pragma unroll
    for(int n=0;n<D_STATE;n++){
      h[n] = __expf(dlt*Arow[n])*h[n] + dx*bt[n];
      y += h[n]*ct[n];
    }
    y += dpd*xv;
    const float z = xz[(size_t)t*(2*D_INNER) + D_INNER + d];
    y *= siluf_(z);
    y_b[(size_t)t*D_INNER + d] = (bf16)y;
  }
}

// ---------------- launcher ----------------
static inline size_t al256(size_t x){ return (x + 255) & ~(size_t)255; }

extern "C" void kernel_launch(void* const* d_in, const int* in_sizes, int n_in,
                              void* d_out, int out_size, void* d_ws, size_t ws_size,
                              hipStream_t stream) {
  const float* x      = (const float*)d_in[0];   // [SEQ, D_MODEL]
  const float* ln_g   = (const float*)d_in[1];
  const float* ln_b   = (const float*)d_in[2];
  const float* Win    = (const float*)d_in[3];   // [2*D_INNER, D_MODEL]
  const float* conv_w = (const float*)d_in[4];   // [D_INNER, 1, D_CONV]
  const float* conv_b = (const float*)d_in[5];
  const float* Wx     = (const float*)d_in[6];   // [NPROJ, D_INNER]
  const float* Wdt    = (const float*)d_in[7];   // [D_INNER, DT_RANK]
  const float* bdt    = (const float*)d_in[8];
  const float* A_log  = (const float*)d_in[9];   // [D_INNER, D_STATE]
  const float* Dp     = (const float*)d_in[10];
  const float* Wout   = (const float*)d_in[11];  // [D_MODEL, D_INNER]
  float* out = (float*)d_out;

  // workspace carve-up (256B aligned)
  char* p = (char*)d_ws; size_t off = 0;
  auto carve = [&](size_t bytes)->char*{ char* r = p + off; off = al256(off + bytes); return r; };
  bf16*  xn_bf   = (bf16*) carve((size_t)SEQ*D_MODEL*2);
  bf16*  win_bf  = (bf16*) carve((size_t)2*D_INNER*D_MODEL*2);
  bf16*  wx_bf   = (bf16*) carve((size_t)NPROJ*D_INNER*2);
  bf16*  wdt_bf  = (bf16*) carve((size_t)D_INNER*DT_RANK*2);
  bf16*  wout_bf = (bf16*) carve((size_t)D_MODEL*D_INNER*2);
  float* xz      = (float*)carve((size_t)SEQ*2*D_INNER*4);
  float* xi_f    = (float*)carve((size_t)SEQ*D_INNER*4);
  bf16*  xi_b    = (bf16*) carve((size_t)SEQ*D_INNER*2);
  float* dbc     = (float*)carve((size_t)SEQ*NPROJ*4);
  bf16*  dt_b    = (bf16*) carve((size_t)SEQ*DT_RANK*2);
  float* delta   = (float*)carve((size_t)SEQ*D_INNER*4);
  float* hend    = (float*)carve((size_t)NCHUNK*D_INNER*D_STATE*4);
  float* hstart  = (float*)carve((size_t)NCHUNK*D_INNER*D_STATE*4);
  float* sdelta  = (float*)carve((size_t)NCHUNK*D_INNER*4);
  bf16*  y_b     = (bf16*) carve((size_t)SEQ*D_INNER*2);
  (void)ws_size; (void)n_in; (void)in_sizes; (void)out_size;

  // weight conversions to bf16
  {
    int n1 = 2*D_INNER*D_MODEL;  k_cvt_bf16<<<(n1+255)/256,256,0,stream>>>(Win,  win_bf,  n1);
    int n2 = NPROJ*D_INNER;      k_cvt_bf16<<<(n2+255)/256,256,0,stream>>>(Wx,   wx_bf,   n2);
    int n3 = D_INNER*DT_RANK;    k_cvt_bf16<<<(n3+255)/256,256,0,stream>>>(Wdt,  wdt_bf,  n3);
    int n4 = D_MODEL*D_INNER;    k_cvt_bf16<<<(n4+255)/256,256,0,stream>>>(Wout, wout_bf, n4);
  }

  // layernorm -> bf16
  k_layernorm_bf16<<<SEQ, 256, 0, stream>>>(x, ln_g, ln_b, xn_bf);

  // xz = xn @ Win^T   (M=2048, N=2048, K=512)
  k_gemm_bf16<EPI_NONE><<<dim3(2*D_INNER/128, SEQ/128), 256, 0, stream>>>(
      xn_bf, D_MODEL, win_bf, D_MODEL, xz, 2*D_INNER,
      SEQ, 2*D_INNER, D_MODEL, nullptr, 0);

  // depthwise causal conv + silu
  k_conv_silu<<<(SEQ*D_INNER)/256, 256, 0, stream>>>(xz, conv_w, conv_b, xi_f, xi_b);

  // dbc = xi @ Wx^T   (M=2048, N=160, K=1024)
  k_gemm_bf16<EPI_NONE><<<dim3((NPROJ+127)/128, SEQ/128), 256, 0, stream>>>(
      xi_b, D_INNER, wx_bf, D_INNER, dbc, NPROJ,
      SEQ, NPROJ, D_INNER, nullptr, 0);

  // dt -> bf16
  k_dt_bf16<<<(SEQ*DT_RANK)/256, 256, 0, stream>>>(dbc, dt_b);

  // delta = softplus(dt @ Wdt^T + bdt)   (M=2048, N=1024, K=32 -> one WMMA step)
  k_gemm_bf16<EPI_SOFTPLUS_BIAS><<<dim3(D_INNER/128, SEQ/128), 256, 0, stream>>>(
      dt_b, DT_RANK, wdt_bf, DT_RANK, delta, D_INNER,
      SEQ, D_INNER, DT_RANK, bdt, 0);

  // chunked selective scan (3 passes)
  k_scan_pass1<<<NCHUNK*(D_INNER/256), 256, 0, stream>>>(delta, xi_f, dbc, A_log, hend, sdelta);
  k_scan_pass2<<<D_INNER/256, 256, 0, stream>>>(hend, sdelta, A_log, hstart);
  k_scan_pass3<<<NCHUNK*(D_INNER/256), 256, 0, stream>>>(delta, xi_f, dbc, A_log, hstart, Dp, xz, y_b);

  // out = y @ Wout^T + x   (M=2048, N=512, K=1024)
  k_gemm_bf16<EPI_RESID><<<dim3(D_MODEL/128, SEQ/128), 256, 0, stream>>>(
      y_b, D_INNER, wout_bf, D_INNER, out, D_MODEL,
      SEQ, D_MODEL, D_INNER, x, D_MODEL);
}
